// MS_StandardConv_27092653703329
// MI455X (gfx1250) — matrix operand order repaired and checked by reading, not compiled, not run on hardware
//
#include <hip/hip_runtime.h>

typedef __attribute__((ext_vector_type(16))) _Float16 v16h;
typedef __attribute__((ext_vector_type(8)))  _Float16 v8h;
typedef __attribute__((ext_vector_type(8)))  float    v8f;
typedef __attribute__((ext_vector_type(4)))  unsigned int u32x4;
typedef __attribute__((ext_vector_type(8)))  int      i32x8;
typedef __attribute__((ext_vector_type(4)))  int      i32x4;

// Problem geometry
#define T_DIM 4
#define B_DIM 8
#define C1    64
#define C2    128
#define H     128
#define W     128
#define NIMG  (T_DIM * B_DIM)          // 32
#define HW    (H * W)                  // 16384
#define ROWE  (W * C1)                 // 8192 elements per NHWC image row
#define K_TOT (C1 * 9)                 // 576;  K order: k = (kh*3+kw)*64 + c1
#define KTILES (K_TOT / 32)            // 18
#define NTILES (C2 / 16)               // 8
#define OUT_ELEMS ((long)NIMG * C2 * HW)       // 67,108,864
#define BN_COUNT  ((float)((long)NIMG * HW))   // 524288 per channel

// Workspace layout (bytes); needs ~67.3 MB
#define BFRAG_ELEMS (K_TOT * C2)       // 73728 f16 -> 147456 B
#define WS_SUM      (BFRAG_ELEMS * 2)  // 147456
#define WS_SUMSQ    (WS_SUM + 512)
#define WS_SPIKE    (WS_SUM + 1024)    // spikes NHWC: 33,554,432 f16 = 64 MiB

// ---------------------------------------------------------------------------
// k0: weights -> WMMA B-fragment layout  Bfrag[ktile][ntile][lane][j] (f16)
//     K order: k = tap*64 + c1  (tap = kh*3+kw).  B k order used here (contiguous, as A):
//     lane<16 col n=lane holds K=0..15 (j), lane>=16 col n=lane-16 K=16..31.
// ---------------------------------------------------------------------------
__global__ void prep_b(const float* __restrict__ conv_w,
                       _Float16* __restrict__ bfrag,
                       float* __restrict__ gsum, float* __restrict__ gsumsq) {
  int e = blockIdx.x * 256 + threadIdx.x;
  if (e < 128) { gsum[e] = 0.0f; gsumsq[e] = 0.0f; }
  if (e >= BFRAG_ELEMS) return;
  int ktile = e >> 12;
  int rem   = e & 4095;
  int ntile = rem >> 9;
  int rem2  = rem & 511;
  int lane  = rem2 >> 4;
  int j     = rem2 & 15;
  int k   = ktile * 32 + ((lane >> 4) << 4) + j;
  int n   = ntile * 16 + (lane & 15);
  int tap = k >> 6;              // kh*3+kw
  int c1  = k & 63;
  bfrag[e] = (_Float16)conv_w[(n * C1 + c1) * 9 + tap];
}

// ---------------------------------------------------------------------------
// k1: LIF recurrence, spikes written NHWC ([img][h][w][c1]) for the TDM path.
//     LDS transpose: read x coalesced along w, write spikes coalesced along c.
// ---------------------------------------------------------------------------
__global__ __launch_bounds__(256) void lif_kernel(const float* __restrict__ x,
                                                  _Float16* __restrict__ s) {
  __shared__ _Float16 tile[C1 * 132];            // padded stride 132 vs 128
  const int tid = threadIdx.x;
  const int b   = blockIdx.x >> 7;
  const int h   = blockIdx.x & 127;
  float mem[32], spike[32];
#pragma unroll
  for (int i = 0; i < 32; ++i) { mem[i] = 0.0f; spike[i] = 0.0f; }

  for (int t = 0; t < T_DIM; ++t) {
    const int  img  = t * B_DIM + b;
    const long xrow = (((long)img * C1) * H + h) * W;     // + c*HW + w
    const long orow = ((long)img * H + h) * ROWE;         // + w*64 + c
#pragma unroll
    for (int i = 0; i < 32; ++i) {
      int e = tid + i * 256;                 // e = c*128 + w
      int c = e >> 7, w = e & 127;
      float xv = x[xrow + (long)c * HW + w];
      mem[i]   = (mem[i] - spike[i] * 0.5f) * 0.25f + xv;
      spike[i] = rintf(fminf(fmaxf(mem[i], 0.0f), 1.0f));
      tile[c * 132 + w] = (_Float16)spike[i];
    }
    __syncthreads();
#pragma unroll
    for (int i = 0; i < 32; ++i) {
      int e = tid + i * 256;                 // e = w*64 + c
      int w = e >> 6, c = e & 63;
      s[orow + e] = tile[c * 132 + w];
    }
    __syncthreads();
  }
}

// ---------------------------------------------------------------------------
// k2: implicit-GEMM 3x3 conv via v_wmma_f32_16x16x32_f16.
//     Block = one (img, h) row.  Spike slab = rows h-1..h+1, NHWC ->
//     one contiguous 48 KB region, staged into LDS by the Tensor Data Mover.
//     A fragment = 2 x ds_load_b128 per lane per k-tile (K = tap*64+c1).
// ---------------------------------------------------------------------------
__global__ __launch_bounds__(256) void conv_wmma(
    const _Float16* __restrict__ sgl,     // spikes NHWC
    const _Float16* __restrict__ bfragg,  // fragment-ready B (L2-resident)
    float* __restrict__ out,
    float* __restrict__ gsum, float* __restrict__ gsumsq) {
  __shared__ __align__(128) _Float16 Sl[3 * ROWE];  // 3*8192 f16 = 49152 B
  __shared__ float red[2 * C2];

  const int tid = threadIdx.x;
  const int img = blockIdx.x >> 7;
  const int h   = blockIdx.x & 127;

  __builtin_prefetch(bfragg, 0, 0);   // global_prefetch_b8: warm shared B

  // ---- stage spike slab rows h-1..h+1 into LDS ----
  const unsigned tile0 = (h == 0) ? (2u * ROWE) : (3u * ROWE);  // elements
  const unsigned tdim0 = (h == 0 || h == 127) ? (2u * ROWE) : (3u * ROWE);
  const int      grow  = (h == 0) ? 0 : (h - 1);
  const _Float16* gp   = sgl + ((long)img * H + grow) * ROWE;

#if __has_builtin(__builtin_amdgcn_tensor_load_to_lds)
  if (tid < 32) {                         // wave 0 issues the TDM op
    unsigned long long ga = (unsigned long long)(const void*)gp;
    unsigned lds = (unsigned)(unsigned long long)(const void*)&Sl[0];
    if (h == 0) lds += 2u * ROWE;         // rows land in Sl rows 1..2 (bytes)

    u32x4 g0;
    g0[0] = 1u;                                          // count=1, user mode
    g0[1] = lds;                                         // lds_addr (bytes)
    g0[2] = (unsigned)(ga & 0xFFFFFFFFu);                // global_addr[31:0]
    g0[3] = (unsigned)((ga >> 32) & 0x01FFFFFFu) | (2u << 30);  // [56:32]+type

    i32x8 g1;
    g1[0] = (int)(1u << 16);                 // data_size=1 (2 bytes)
    g1[1] = (int)((tdim0 & 0xFFFFu) << 16);  // tensor_dim0[15:0]
    g1[2] = (int)((tdim0 >> 16) | (1u << 16));   // tensor_dim0[31:16], tdim1=1
    g1[3] = (int)(tile0 << 16);              // tile_dim0 (16b, <=24576)
    g1[4] = 1;                               // tile_dim1=1, tile_dim2=0
    g1[5] = (int)tdim0;                      // tensor_dim0_stride[31:0]
    g1[6] = 0;
    g1[7] = 0;
    i32x4 gz = {0, 0, 0, 0};
#if __clang_major__ >= 23
    i32x8 gz8 = {0, 0, 0, 0, 0, 0, 0, 0};
    __builtin_amdgcn_tensor_load_to_lds(g0, g1, gz, gz, gz8, 0);
#else
    __builtin_amdgcn_tensor_load_to_lds(g0, g1, gz, gz, 0);
#endif
    __builtin_amdgcn_s_wait_tensorcnt(0);
  }
  if (h == 0) {                             // zero Sl row 0 (above h-1)
    v8h z = {};
    for (int e = tid * 8; e < ROWE; e += 256 * 8) *(v8h*)&Sl[e] = z;
  }
#else
  // Fallback: cooperative copy with row clamp.
  for (int e = tid; e < 3 * ROWE; e += 256) {
    int r = e / ROWE, idx = e - r * ROWE;
    int row = h - 1 + r;
    Sl[e] = ((unsigned)row < (unsigned)H)
                ? sgl[((long)img * H + row) * ROWE + idx] : (_Float16)0.0f;
  }
#endif
  __syncthreads();

  const int wave = tid >> 5;
  const int lane = tid & 31;
  const int half = lane >> 4;
  const int mrow = lane & 15;
  const int wb   = wave * 16 + mrow;        // output w of this lane's A row

  v8f acc[NTILES];
#pragma unroll
  for (int nt = 0; nt < NTILES; ++nt) acc[nt] = (v8f){0, 0, 0, 0, 0, 0, 0, 0};

#pragma unroll
  for (int kt = 0; kt < KTILES; ++kt) {
    // A k order used here (contiguous, as B): element chunks j0..7 -> K base kt*32 + half*16,
    //                     j8..15 -> K base kt*32 + half*16 + 8.
    v8h ch[2];
#pragma unroll
    for (int c = 0; c < 2; ++c) {
      int kbase = kt * 32 + half * 16 + c * 8;  // constant tap/c0 after unroll
      int tap = kbase >> 6;                     // kh*3+kw
      int c0  = kbase & 63;
      int kh  = tap / 3;
      int kw  = tap - kh * 3;
      int wi  = wb + kw - 1;                    // input column (halo)
      v8h z = {};
      ch[c] = ((unsigned)wi < (unsigned)W)
                  ? *(const v8h*)&Sl[kh * ROWE + wi * C1 + c0] : z;
    }
    v16h a = __builtin_shufflevector(ch[0], ch[1], 0, 1, 2, 3, 4, 5, 6, 7,
                                     8, 9, 10, 11, 12, 13, 14, 15);
#pragma unroll
    for (int nt = 0; nt < NTILES; ++nt) {
      v16h b = *(const v16h*)(bfragg + (((kt * NTILES + nt) * 32 + lane) << 4));
      acc[nt] = __builtin_amdgcn_wmma_f32_16x16x32_f16(
          false, a, false, b, (short)0, acc[nt], false, false);
    }
  }

  // ---- epilogue: store y (NCHW), per-channel BN sums ----
  float ls1[NTILES], ls2[NTILES];
  const long obase = (long)img * C2 * HW + (long)h * W;
#pragma unroll
  for (int nt = 0; nt < NTILES; ++nt) {
    int c2 = nt * 16 + mrow;
    float s1 = 0.0f, s2 = 0.0f;
#pragma unroll
    for (int v = 0; v < 8; ++v) {
      float y  = acc[nt][v];
      int   wo = wave * 16 + half * 8 + v;   // C/D layout: M = v + half*8
      out[obase + (long)c2 * HW + wo] = y;
      s1 += y;
      s2 += y * y;
    }
    ls1[nt] = s1; ls2[nt] = s2;
  }
  __syncthreads();
  red[tid] = 0.0f;
  __syncthreads();
#pragma unroll
  for (int nt = 0; nt < NTILES; ++nt) {
    int c2 = nt * 16 + mrow;
    atomicAdd(&red[c2], ls1[nt]);
    atomicAdd(&red[C2 + c2], ls2[nt]);
  }
  __syncthreads();
  if (tid < C2) {
    atomicAdd(&gsum[tid], red[tid]);
    atomicAdd(&gsumsq[tid], red[C2 + tid]);
  }
}

// ---------------------------------------------------------------------------
// k3: BatchNorm2d (training stats) applied in place on d_out.
// ---------------------------------------------------------------------------
__global__ void bn_kernel(float* __restrict__ out,
                          const float* __restrict__ gsum,
                          const float* __restrict__ gsumsq,
                          const float* __restrict__ gamma,
                          const float* __restrict__ beta) {
  long idx = (long)blockIdx.x * 256 + threadIdx.x;
  int c2 = (int)((idx >> 14) & (C2 - 1));
  float inv  = 1.0f / BN_COUNT;
  float mean = gsum[c2] * inv;
  float var  = gsumsq[c2] * inv - mean * mean;
  float rstd = rsqrtf(var + 1e-5f);
  out[idx] = (out[idx] - mean) * rstd * gamma[c2] + beta[c2];
}

// ---------------------------------------------------------------------------
extern "C" void kernel_launch(void* const* d_in, const int* in_sizes, int n_in,
                              void* d_out, int out_size, void* d_ws, size_t ws_size,
                              hipStream_t stream) {
  (void)in_sizes; (void)n_in; (void)out_size; (void)ws_size;
  const float* x      = (const float*)d_in[0];
  const float* conv_w = (const float*)d_in[1];
  const float* gamma  = (const float*)d_in[2];
  const float* beta   = (const float*)d_in[3];
  float* out = (float*)d_out;

  char* ws = (char*)d_ws;
  _Float16* bfrag  = (_Float16*)(ws);
  float*    gsum   = (float*)(ws + WS_SUM);
  float*    gsumsq = (float*)(ws + WS_SUMSQ);
  _Float16* sbuf   = (_Float16*)(ws + WS_SPIKE);

  prep_b<<<(BFRAG_ELEMS + 255) / 256, 256, 0, stream>>>(conv_w, bfrag, gsum, gsumsq);
  lif_kernel<<<B_DIM * H, 256, 0, stream>>>(x, sbuf);
  conv_wmma<<<NIMG * H, 256, 0, stream>>>(sbuf, bfrag, out, gsum, gsumsq);
  bn_kernel<<<(int)(OUT_ELEMS / 256), 256, 0, stream>>>(out, gsum, gsumsq, gamma, beta);
}
